// CompresSAEEncoder_6657199309556
// MI455X (gfx1250) — matrix operand order repaired and checked by reading, not compiled, not run on hardware
//
#include <hip/hip_runtime.h>
#include <hip/hip_bf16.h>

// ---- trivially-copyable vector types (safe in unions, lower to b128 ops) ----
typedef __attribute__((ext_vector_type(16))) __bf16         v16bf;
typedef __attribute__((ext_vector_type(8)))  float          v8f;
typedef __attribute__((ext_vector_type(4)))  float          f32x4;
typedef __attribute__((ext_vector_type(4)))  unsigned int   u32x4;
typedef __attribute__((ext_vector_type(8)))  unsigned int   u32x8;
typedef __attribute__((ext_vector_type(4)))  unsigned short u16x4;

#define TPB 256
#define BM 128          // workgroup tile rows
#define BN 256          // workgroup tile cols
#define BK 32           // K step (one bf16 WMMA K)
#define PA 40           // LDS pitch (bf16) for As rows: 32 + 8 pad (bank spread, keeps 16B align)
#define PB 40           // LDS pitch (bf16) for Bs cols (stored transposed [n][k]); even -> dword align

static __device__ inline v16bf make_frag(u32x4 lo, u32x4 hi) {
  u32x8 w = __builtin_shufflevector(lo, hi, 0, 1, 2, 3, 4, 5, 6, 7);
  union { u32x8 u; v16bf v; } f;
  f.u = w;
  return f.v;
}

// ---------------- Kernel 1: per-row inverse L2 norm ----------------
__global__ __launch_bounds__(TPB) void rownorm_kernel(const float* __restrict__ x,
                                                      float* __restrict__ inv_norm,
                                                      int din) {
  __shared__ float red[TPB];
  const int row = blockIdx.x;
  const float* xr = x + (size_t)row * din;
  float acc = 0.0f;
  for (int i = threadIdx.x; i < din; i += TPB) {
    float v = xr[i];
    acc += v * v;
  }
  red[threadIdx.x] = acc;
  __syncthreads();
  for (int s = TPB / 2; s > 0; s >>= 1) {
    if (threadIdx.x < s) red[threadIdx.x] += red[threadIdx.x + s];
    __syncthreads();
  }
  if (threadIdx.x == 0) inv_norm[row] = 1.0f / sqrtf(red[0]);
}

// ---------------- Kernel 2: e = (x * inv_norm) @ W + b  (bf16 WMMA, f32 acc) ----------------
// 8 waves as 2(m) x 4(n); each wave owns a 64x64 sub-tile = 4x4 WMMA fragments
// (16 WMMAs per K-step per wave; 0.5KB LDS reads per WMMA).
// Grid (B/BM, DEMB/BN): blockIdx.x fastest => 64 M-blocks consume one 4MB W panel
// from L2 while x (128MB) becomes L2-resident across N-panels.
__global__ __launch_bounds__(TPB) void gemm_bf16_wmma_kernel(
    const float* __restrict__ x, const float* __restrict__ W,
    const float* __restrict__ bias, const float* __restrict__ inv_norm,
    float* __restrict__ out, int din, int demb) {
  __shared__ __align__(16) __bf16 As[BM * PA];   // [m][k]
  __shared__ __align__(16) __bf16 Bs[BN * PB];   // [n][k] (transposed)

  const int tid  = threadIdx.x;
  const int lane = tid & 31;
  const int wave = tid >> 5;
  const int wm = (wave & 1) * 64;     // wave sub-tile rows (2 groups)
  const int wn = (wave >> 1) * 64;    // wave sub-tile cols (4 groups)

  const int mb = blockIdx.x * BM;
  const int nb = blockIdx.y * BN;

  // Fixed per-thread staging coordinates (hoisted out of K loop).
  // A: 4 rounds of float4 -> 128 rows x 32 k
  int arow[4], ak[4];
  float ascale[4];
  const float* aptr[4];
  // B: 4 rounds; each thread takes a k-pair (2 rows) x 4 cols -> packs dword stores
  int bkp[4], bn4[4];
#pragma unroll
  for (int r = 0; r < 4; ++r) {
    int idx = r * TPB + tid;        // 0..1023
    arow[r] = idx >> 3;             // 0..127
    ak[r]   = (idx & 7) << 2;       // 0..28 step 4
    ascale[r] = inv_norm[mb + arow[r]];
    aptr[r] = x + (size_t)(mb + arow[r]) * din;
    bkp[r]  = (idx & 15) << 1;      // k-pair base: 0,2,..,30
    bn4[r]  = (idx >> 4) << 2;      // 0..252 step 4
  }

  v8f acc[4][4];
#pragma unroll
  for (int i = 0; i < 4; ++i)
#pragma unroll
    for (int j = 0; j < 4; ++j)
#pragma unroll
      for (int e = 0; e < 8; ++e) acc[i][j][e] = 0.0f;

  for (int kb = 0; kb < din; kb += BK) {
    // ---- stage A: x rows scaled to unit norm, f32 -> bf16, b128 global / b64 ds
#pragma unroll
    for (int r = 0; r < 4; ++r) {
      f32x4 v = *(const f32x4*)(aptr[r] + kb + ak[r]);
      float s = ascale[r];
      union { __bf16 h[4]; u16x4 u; } pk;
      pk.h[0] = (__bf16)(v.x * s);
      pk.h[1] = (__bf16)(v.y * s);
      pk.h[2] = (__bf16)(v.z * s);
      pk.h[3] = (__bf16)(v.w * s);
      *(u16x4*)&As[arow[r] * PA + ak[r]] = pk.u;
    }
    // ---- stage B: two adjacent k-rows, pack k-pairs -> 4 x ds_store_b32 (transposed [n][k])
#pragma unroll
    for (int r = 0; r < 4; ++r) {
      const float* w0 = W + (size_t)(kb + bkp[r]) * demb + nb + bn4[r];
      f32x4 vk0 = *(const f32x4*)(w0);
      f32x4 vk1 = *(const f32x4*)(w0 + demb);
      union { __bf16 h[2]; unsigned int u; } pk;
#pragma unroll
      for (int c = 0; c < 4; ++c) {
        float lo = (c == 0) ? vk0.x : (c == 1) ? vk0.y : (c == 2) ? vk0.z : vk0.w;
        float hi = (c == 0) ? vk1.x : (c == 1) ? vk1.y : (c == 2) ? vk1.z : vk1.w;
        pk.h[0] = (__bf16)lo;
        pk.h[1] = (__bf16)hi;
        *(unsigned int*)&Bs[(bn4[r] + c) * PB + bkp[r]] = pk.u;
      }
    }
    __syncthreads();

    // ---- fragment loads in WMMA-native layouts (ISA 7.12.2)
    const int m0  = wm + (lane & 15);
    const int ka  = (lane >> 4) << 3;   // 0 or 8   (A: K chunks {ka..ka+7, ka+16..ka+23})
    const int n0  = wn + (lane & 15);
    const int kb0 = (lane >> 4) << 3;   // 0 or 8   (B: K chunks {kb0..kb0+7, kb0+16..kb0+23})
    v16bf afrag[4], bfrag[4];
#pragma unroll
    for (int i = 0; i < 4; ++i) {
      const __bf16* p = &As[(m0 + i * 16) * PA + ka];
      afrag[i] = make_frag(*(const u32x4*)(p), *(const u32x4*)(p + 16));
    }
#pragma unroll
    for (int j = 0; j < 4; ++j) {
      const __bf16* p = &Bs[(n0 + j * 16) * PB + kb0];
      bfrag[j] = make_frag(*(const u32x4*)(p), *(const u32x4*)(p + 16));
    }
#pragma unroll
    for (int i = 0; i < 4; ++i)
#pragma unroll
      for (int j = 0; j < 4; ++j)
        acc[i][j] = __builtin_amdgcn_wmma_f32_16x16x32_bf16(
            false, afrag[i], false, bfrag[j], (short)0, acc[i][j], false, false);
    __syncthreads();
  }

  // ---- epilogue: D layout -> row = base + r + 8*(lane>=16), col = lane&15
  const int rbase = (lane >> 4) << 3;
#pragma unroll
  for (int j = 0; j < 4; ++j) {
    const int col = nb + wn + j * 16 + (lane & 15);
    const float bv = bias[col];
#pragma unroll
    for (int i = 0; i < 4; ++i) {
      const int row0 = mb + wm + i * 16 + rbase;
      float* o = out + (size_t)row0 * demb + col;
#pragma unroll
      for (int r = 0; r < 8; ++r) o[(size_t)r * demb] = acc[i][j][r] + bv;
    }
  }
}

// ---------------- Kernel 3: per-row exact top-k mask via 4x8-bit radix select ----------------
// One workgroup per row; row (128KB) lives in dynamic LDS (<=320KB/WG on CDNA5).
__global__ __launch_bounds__(TPB) void topk_mask_kernel(float* __restrict__ e,
                                                        const int* __restrict__ kptr,
                                                        int demb) {
  extern __shared__ unsigned char smem[];
  float* rowv = (float*)smem;                                        // demb floats
  unsigned* hist = (unsigned*)(smem + (size_t)demb * sizeof(float)); // 256 bins
  unsigned* ctrl = hist + 256;   // [0]=prefix, [1]=remaining, [2]=tie counter

  const int tid = threadIdx.x;
  float* erow = e + (size_t)blockIdx.x * demb;

  for (int i = tid; i < demb; i += TPB) rowv[i] = erow[i];
  if (tid == 0) { ctrl[0] = 0u; ctrl[1] = (unsigned)(*kptr); ctrl[2] = 0u; }
  __syncthreads();

  // Radix-select the k-th largest |e| (non-negative float bits order like uints).
  for (int pass = 0; pass < 4; ++pass) {
    const int shift = 24 - 8 * pass;
    hist[tid & 255] = 0u;
    __syncthreads();
    const unsigned prefix = ctrl[0];
    for (int i = tid; i < demb; i += TPB) {
      unsigned key = __float_as_uint(rowv[i]) & 0x7fffffffu;
      bool in = (pass == 0) || ((key >> (shift + 8)) == (prefix >> (shift + 8)));
      if (in) atomicAdd(&hist[(key >> shift) & 255u], 1u);
    }
    __syncthreads();
    if (tid == 0) {
      unsigned rem = ctrl[1];
      unsigned pfx = ctrl[0];
      for (int bkt = 255; bkt >= 0; --bkt) {
        unsigned c = hist[bkt];
        if (c >= rem) { ctrl[0] = pfx | ((unsigned)bkt << shift); break; }
        rem -= c;
      }
      ctrl[1] = rem;
    }
    __syncthreads();
  }

  const unsigned t = ctrl[0];      // exact k-th largest |e| (bit pattern)
  const unsigned quota = ctrl[1];  // how many ==t values to keep
  for (int i = tid; i < demb; i += TPB) {
    float v = rowv[i];
    unsigned key = __float_as_uint(v) & 0x7fffffffu;
    float o = 0.0f;
    if (key > t) {
      o = v;
    } else if (key == t) {
      unsigned p = atomicAdd(&ctrl[2], 1u);
      if (p < quota) o = v;
    }
    erow[i] = o;   // in-place: dense e -> masked output
  }
}

extern "C" void kernel_launch(void* const* d_in, const int* in_sizes, int n_in,
                              void* d_out, int out_size, void* d_ws, size_t ws_size,
                              hipStream_t stream) {
  const float* x = (const float*)d_in[0];
  const float* W = (const float*)d_in[1];
  const float* b = (const float*)d_in[2];
  const int* kptr = (const int*)d_in[3];

  const int demb  = in_sizes[2];            // 32768
  const int din   = in_sizes[1] / demb;     // 4096
  const int Brows = in_sizes[0] / din;      // 8192

  float* inv_norm = (float*)d_ws;           // 32KB scratch
  float* out = (float*)d_out;

  rownorm_kernel<<<Brows, TPB, 0, stream>>>(x, inv_norm, din);

  dim3 grid(Brows / BM, demb / BN);         // x fastest: M-blocks share the W N-panel in L2
  gemm_bf16_wmma_kernel<<<grid, TPB, 0, stream>>>(x, W, b, inv_norm, out, din, demb);

  size_t smem3 = (size_t)demb * sizeof(float) + (256 + 8) * sizeof(unsigned);
  topk_mask_kernel<<<Brows, TPB, smem3, stream>>>(out, kptr, demb);
}
